// GRU_70858370449499
// MI455X (gfx1250) — compile-verified, matrix operand order checked by reading, not run on hardware
//
#include <hip/hip_runtime.h>
#include <hip/hip_bf16.h>
#include <stdint.h>

// ---------------------------------------------------------------------------
// Problem constants (from reference): B=64, I=T=H=512, L=4, O=128
// ---------------------------------------------------------------------------
#define GRU_B   64
#define GRU_T   512
#define GRU_H   512
#define GRU_G3  1536          // 3*H
#define GRU_O   128
#define GRU_L   4
#define GRU_M   (GRU_B * GRU_T)   // 32768 rows of the big GEMMs

typedef __bf16 bf16_t;
typedef __bf16 v16bf __attribute__((ext_vector_type(16)));
typedef float  v8f   __attribute__((ext_vector_type(8)));
typedef unsigned int v4u  __attribute__((ext_vector_type(4)));
typedef int          v8ii __attribute__((ext_vector_type(8)));
typedef int          v4ii __attribute__((ext_vector_type(4)));

union FragB16 { v16bf v; uint4 u[2]; };

__device__ __forceinline__ v8f wmma_bf16(v16bf a, v16bf b, v8f c) {
    // D(f32 16x16) = A(16x32 bf16) * B(32x16 bf16) + C
    return __builtin_amdgcn_wmma_f32_16x16x32_bf16(
        /*neg_a=*/false, a, /*neg_b=*/false, b,
        /*c_mod=*/(short)0, c, /*reuse_a=*/false, /*reuse_b=*/false);
}

__device__ __forceinline__ float sigmoidf_(float x) {
    return 1.0f / (1.0f + __expf(-x));
}

// ---------------------------------------------------------------------------
// Tensor Data Mover: 2D bf16 tile (tile_d0 elems along K, tile_d1 rows) from
// a row-major [*, d0_stride] bf16 tensor into LDS (packed row-major).
// Descriptor bitfields per CDNA5 ISA.
// This toolchain uses the 6-arg builtin: (v4u, v8i, v4i, v4i, v8i, i32 cpol).
// ---------------------------------------------------------------------------
__device__ __forceinline__ void tdm_load_tile_2d_bf16(
    unsigned lds_off, const void* gaddr,
    unsigned tensor_d0, unsigned tensor_d1,
    unsigned tile_d0, unsigned tile_d1, unsigned d0_stride)
{
    unsigned long long ga = (unsigned long long)(uintptr_t)gaddr;
    v4u g0;
    g0[0] = 1u;                                   // count=1, user descriptor
    g0[1] = lds_off;                              // LDS byte address
    g0[2] = (unsigned)ga;                         // global_addr[31:0]
    g0[3] = (unsigned)((ga >> 32) & 0x01FFFFFFull) | (2u << 30);  // addr[56:32], type=2
    v8ii g1;
    g1[0] = (int)(1u << 16);                      // data_size=1 (2 bytes), no multicast
    g1[1] = (int)((tensor_d0 & 0xFFFFu) << 16);   // tensor_dim0[15:0]
    g1[2] = (int)(((tensor_d0 >> 16) & 0xFFFFu) | ((tensor_d1 & 0xFFFFu) << 16));
    g1[3] = (int)(((tensor_d1 >> 16) & 0xFFFFu) | ((tile_d0 & 0xFFFFu) << 16));
    g1[4] = (int)(tile_d1 & 0xFFFFu);             // tile_dim1 (tile_dim2=0)
    g1[5] = (int)d0_stride;                       // tensor_dim0_stride[31:0]
    g1[6] = 0;                                    // stride[47:32]=0, dim1_stride lo=0
    g1[7] = 0;
    v4ii g2 = {0, 0, 0, 0};                       // 2D tensor: groups 2/3 unused
    v4ii g3 = {0, 0, 0, 0};
    v8ii g4 = {0, 0, 0, 0, 0, 0, 0, 0};           // extra group (clang-23 form)
    __builtin_amdgcn_tensor_load_to_lds(g0, g1, g2, g3, g4, 0);
}

// ---------------------------------------------------------------------------
// Kernel: transpose input [B][I][T] fp32 -> x0 [B][T][I] fp32 (LDS-tiled)
// ---------------------------------------------------------------------------
__global__ __launch_bounds__(256) void gru_transpose_kernel(
    const float* __restrict__ in, float* __restrict__ out)
{
    __shared__ float tile[32][33];
    const int b  = blockIdx.z;
    const int t0 = blockIdx.x * 32;
    const int i0 = blockIdx.y * 32;
    const int tx = threadIdx.x;       // 0..31
    const int ty = threadIdx.y;       // 0..7
    const float* src = in + ((size_t)b * GRU_H + i0) * GRU_T + t0;
#pragma unroll
    for (int k = 0; k < 32; k += 8)
        tile[ty + k][tx] = src[(size_t)(ty + k) * GRU_T + tx];
    __syncthreads();
    float* dst = out + ((size_t)b * GRU_T + t0) * GRU_H + i0;
#pragma unroll
    for (int k = 0; k < 32; k += 8)
        dst[(size_t)(ty + k) * GRU_H + tx] = tile[tx][ty + k];
}

// ---------------------------------------------------------------------------
// Kernel: fp32 -> bf16 weight conversion (grid-stride)
// ---------------------------------------------------------------------------
__global__ __launch_bounds__(256) void gru_cvt_bf16_kernel(
    const float* __restrict__ src, bf16_t* __restrict__ dst, int n)
{
    for (int i = blockIdx.x * 256 + threadIdx.x; i < n; i += gridDim.x * 256)
        dst[i] = (bf16_t)src[i];
}

// ---------------------------------------------------------------------------
// Kernel: zero the inter-workgroup barrier counters
// ---------------------------------------------------------------------------
__global__ void gru_init_ws_kernel(unsigned* cnts)
{
    if (threadIdx.x < 16) cnts[threadIdx.x] = 0u;
}

// ---------------------------------------------------------------------------
// GEMM + bias:  C[M][Ntot] = X[M][512] * Wb[Ntot][512]^T + bias[Ntot]
// Tiles: BM=128, BN=128, BK=32.  8 waves; wave (wr,wc) owns 64x32.
// W tile staged into LDS by the Tensor Data Mover; X converted fp32->bf16.
// ---------------------------------------------------------------------------
__global__ __launch_bounds__(256) void gru_gemm_bias_kernel(
    const float*  __restrict__ X,
    const bf16_t* __restrict__ Wb,
    const float*  __restrict__ bias,
    float*        __restrict__ Cout,
    int Ntot)
{
    __shared__ __attribute__((aligned(16))) bf16_t Xs[128][32];
    __shared__ __attribute__((aligned(16))) bf16_t Ws[128][32];
    const int K  = 512;
    const int n0 = blockIdx.x * 128;
    const int m0 = blockIdx.y * 128;
    const int tid   = threadIdx.x;
    const int lane  = tid & 31;
    const int wave  = tid >> 5;
    const int wr    = wave >> 2;      // 0..1 -> rows wr*64
    const int wc    = wave & 3;       // 0..3 -> cols wc*32
    const int mlane = lane & 15;
    const int glane = lane >> 4;

    const unsigned ws_lds = (unsigned)(uintptr_t)(&Ws[0][0]);

    v8f acc[4][2];
#pragma unroll
    for (int i = 0; i < 4; ++i)
#pragma unroll
        for (int j = 0; j < 2; ++j)
            acc[i][j] = (v8f){0.f, 0.f, 0.f, 0.f, 0.f, 0.f, 0.f, 0.f};

    const int xrow  = tid >> 1;       // 0..127
    const int xhalf = tid & 1;        // 0..1

    for (int k0 = 0; k0 < K; k0 += 32) {
        // --- async TDM: W tile rows n0..n0+127, K window k0..k0+31 -> Ws ---
        if (tid < 32) {
            tdm_load_tile_2d_bf16(ws_lds, Wb + (size_t)n0 * K + k0,
                                  /*tensor_d0=*/K, /*tensor_d1=*/128,
                                  /*tile_d0=*/32, /*tile_d1=*/128,
                                  /*d0_stride=*/K);
        }
        // --- X tile: 128x32 fp32 -> bf16, cooperative ---
        {
            const float4* s = (const float4*)(X + (size_t)(m0 + xrow) * K + k0 + xhalf * 16);
            float4 f0 = s[0], f1 = s[1], f2 = s[2], f3 = s[3];
            union { bf16_t h[8]; uint4 u; } p0, p1;
            p0.h[0] = (bf16_t)f0.x; p0.h[1] = (bf16_t)f0.y; p0.h[2] = (bf16_t)f0.z; p0.h[3] = (bf16_t)f0.w;
            p0.h[4] = (bf16_t)f1.x; p0.h[5] = (bf16_t)f1.y; p0.h[6] = (bf16_t)f1.z; p0.h[7] = (bf16_t)f1.w;
            p1.h[0] = (bf16_t)f2.x; p1.h[1] = (bf16_t)f2.y; p1.h[2] = (bf16_t)f2.z; p1.h[3] = (bf16_t)f2.w;
            p1.h[4] = (bf16_t)f3.x; p1.h[5] = (bf16_t)f3.y; p1.h[6] = (bf16_t)f3.z; p1.h[7] = (bf16_t)f3.w;
            *(uint4*)&Xs[xrow][xhalf * 16]     = p0.u;
            *(uint4*)&Xs[xrow][xhalf * 16 + 8] = p1.u;
        }
        if (tid < 32) __builtin_amdgcn_s_wait_tensorcnt(0);
        __syncthreads();

        FragB16 afr[4], bfr[2];
#pragma unroll
        for (int i = 0; i < 4; ++i) {
            const int r = wr * 64 + i * 16 + mlane;
            afr[i].u[0] = *(const uint4*)&Xs[r][glane * 8];
            afr[i].u[1] = *(const uint4*)&Xs[r][16 + glane * 8];
        }
#pragma unroll
        for (int j = 0; j < 2; ++j) {
            const int r = wc * 32 + j * 16 + mlane;
            bfr[j].u[0] = *(const uint4*)&Ws[r][glane * 8];
            bfr[j].u[1] = *(const uint4*)&Ws[r][16 + glane * 8];
        }
#pragma unroll
        for (int i = 0; i < 4; ++i)
#pragma unroll
            for (int j = 0; j < 2; ++j)
                acc[i][j] = wmma_bf16(afr[i].v, bfr[j].v, acc[i][j]);
        __syncthreads();
    }

    // Epilogue: C element v of lane <-> row glane*8+v, col mlane
#pragma unroll
    for (int i = 0; i < 4; ++i) {
#pragma unroll
        for (int j = 0; j < 2; ++j) {
            const int mbase = m0 + wr * 64 + i * 16 + glane * 8;
            const int ncol  = n0 + wc * 32 + j * 16 + mlane;
            const float bv  = bias[ncol];
#pragma unroll
            for (int v = 0; v < 8; ++v)
                Cout[(size_t)(mbase + v) * Ntot + ncol] = acc[i][j][v] + bv;
        }
    }
}

// ---------------------------------------------------------------------------
// Inter-workgroup barrier within a batch-group (4 WGs), monotonic counter.
// ---------------------------------------------------------------------------
__device__ __forceinline__ void group_barrier(unsigned* cnt, unsigned target)
{
    __syncthreads();
    if (threadIdx.x == 0) {
        __threadfence();  // publish h stores before arriving
        __hip_atomic_fetch_add(cnt, 1u, __ATOMIC_RELEASE, __HIP_MEMORY_SCOPE_AGENT);
        while (__hip_atomic_load(cnt, __ATOMIC_ACQUIRE, __HIP_MEMORY_SCOPE_AGENT) < target)
            __builtin_amdgcn_s_sleep(1);
    }
    __syncthreads();
}

// ---------------------------------------------------------------------------
// GRU recurrence for one layer.
// Grid (4 j-chunks, 4 batch-groups) x 256 threads.  Each batch-group owns a
// 16-row h state (bf16 in global, re-staged to LDS each step).  Each WG owns
// a 128-wide hidden-unit chunk: per wave one 16-wide j-subtile, 3 gate tiles.
// ---------------------------------------------------------------------------
__global__ __launch_bounds__(256) void gru_recurrence_kernel(
    const float*  __restrict__ xw,     // [B*T][1536] = X*Wih^T + bih
    const bf16_t* __restrict__ WhhB,   // [1536][512] bf16 (this layer)
    const float*  __restrict__ bhh,    // [1536]
    float*        __restrict__ xout,   // [B*T][512] layer output
    bf16_t*       __restrict__ hbf,    // [4 groups][16][512] bf16 h state
    unsigned*     __restrict__ barrier_cnt)  // [4] this layer's counters
{
    __shared__ __attribute__((aligned(16))) bf16_t hstage[16][512]; // 16 KB
    __shared__ float hloc[16][128];                                 // 8 KB (own fp32 h)
    const int K = 512;
    const int bgroup = blockIdx.y;        // 0..3
    const int wgJ    = blockIdx.x;        // 0..3
    const int tid    = threadIdx.x;
    const int lane   = tid & 31;
    const int wave   = tid >> 5;
    const int mlane  = lane & 15;
    const int glane  = lane >> 4;
    unsigned* cnt = barrier_cnt + bgroup;
    bf16_t* hbf_g = hbf + (size_t)bgroup * 16 * 512;
    const int j0   = wgJ * 128 + wave * 16;     // wave's global j base
    const int jcol = j0 + mlane;                // this lane's hidden unit

    // Precompute B-matrix (Whh row) pointers for the 3 gates
    const bf16_t* brow[3];
#pragma unroll
    for (int g = 0; g < 3; ++g)
        brow[g] = WhhB + (size_t)(g * 512 + jcol) * K;

    // h := 0 (local fp32 chunk + own slice of the group's bf16 state)
    for (int i = tid; i < 16 * 128; i += 256) ((float*)hloc)[i] = 0.f;
    for (int i = tid; i < 16 * 128; i += 256) {
        const int r = i >> 7, c = i & 127;
        hbf_g[r * 512 + wgJ * 128 + c] = (bf16_t)0.f;
    }
    unsigned phase = 1;
    group_barrier(cnt, 4u * phase);

    const float bhr = bhh[jcol];
    const float bhz = bhh[512 + jcol];
    const float bhn = bhh[1024 + jcol];

    for (int t = 0; t < GRU_T; ++t) {
        // Stage the group's full 16x512 bf16 h into LDS (shared A-matrix)
        {
            const uint4* s = (const uint4*)hbf_g;
            uint4* d = (uint4*)&hstage[0][0];
            for (int i = tid; i < (16 * 512 * 2) / 16; i += 256) d[i] = s[i];
        }
        __syncthreads();

        v8f acc[3];
#pragma unroll
        for (int g = 0; g < 3; ++g)
            acc[g] = (v8f){0.f, 0.f, 0.f, 0.f, 0.f, 0.f, 0.f, 0.f};

#pragma unroll
        for (int ks = 0; ks < 16; ++ks) {
            const int k0 = ks * 32;
            FragB16 a;
            a.u[0] = *(const uint4*)&hstage[mlane][k0 + glane * 8];
            a.u[1] = *(const uint4*)&hstage[mlane][k0 + 16 + glane * 8];
#pragma unroll
            for (int g = 0; g < 3; ++g) {
                FragB16 b;
                b.u[0] = *(const uint4*)(brow[g] + k0 + glane * 8);
                b.u[1] = *(const uint4*)(brow[g] + k0 + 16 + glane * 8);
                acc[g] = wmma_bf16(a.v, b.v, acc[g]);
            }
        }

        // Gate math (fp32):  r,z = sigmoid; n = tanh; h' = (1-z)*n + z*h
#pragma unroll
        for (int v = 0; v < 8; ++v) {
            const int brw = glane * 8 + v;                       // local batch row
            const size_t xrow = ((size_t)(bgroup * 16 + brw) * GRU_T + t) * GRU_G3;
            const float xr = xw[xrow + jcol];
            const float xz = xw[xrow + 512 + jcol];
            const float xn = xw[xrow + 1024 + jcol];
            const float hp = hloc[brw][wave * 16 + mlane];
            const float r  = sigmoidf_(xr + acc[0][v] + bhr);
            const float z  = sigmoidf_(xz + acc[1][v] + bhz);
            const float n  = tanhf(xn + r * (acc[2][v] + bhn));
            const float hn = (1.0f - z) * n + z * hp;
            hloc[brw][wave * 16 + mlane] = hn;
            hbf_g[brw * 512 + jcol] = (bf16_t)hn;
            xout[((size_t)(bgroup * 16 + brw) * GRU_T + t) * GRU_H + jcol] = hn;
        }

        ++phase;
        group_barrier(cnt, 4u * phase);
    }
}

// ---------------------------------------------------------------------------
// Host launcher
// ---------------------------------------------------------------------------
extern "C" void kernel_launch(void* const* d_in, const int* in_sizes, int n_in,
                              void* d_out, int out_size, void* d_ws, size_t ws_size,
                              hipStream_t stream)
{
    const float* inp = (const float*)d_in[0];   // [B][I][T]
    const float* Wih = (const float*)d_in[1];   // [L][1536][512]
    const float* Whh = (const float*)d_in[2];   // [L][1536][512]
    const float* bih = (const float*)d_in[3];   // [L][1536]
    const float* bhh = (const float*)d_in[4];   // [L][1536]
    const float* fcw = (const float*)d_in[5];   // [128][512]
    const float* fcb = (const float*)d_in[6];   // [128]

    char* base = (char*)d_ws;
    size_t off = 0;
    auto alloc = [&](size_t bytes) -> void* {
        off = (off + 255) & ~(size_t)255;
        void* p = base + off;
        off += bytes;
        return p;
    };
    float*    xA    = (float*)alloc((size_t)GRU_M * GRU_H * 4);
    float*    xB    = (float*)alloc((size_t)GRU_M * GRU_H * 4);
    float*    xwbuf = (float*)alloc((size_t)GRU_M * GRU_G3 * 4);
    bf16_t*   WihB  = (bf16_t*)alloc((size_t)GRU_L * GRU_G3 * GRU_H * 2);
    bf16_t*   WhhB  = (bf16_t*)alloc((size_t)GRU_L * GRU_G3 * GRU_H * 2);
    bf16_t*   fcwB  = (bf16_t*)alloc((size_t)GRU_O * GRU_H * 2);
    bf16_t*   hbf   = (bf16_t*)alloc((size_t)4 * 16 * GRU_H * 2);
    unsigned* cnts  = (unsigned*)alloc(16 * sizeof(unsigned));

    gru_init_ws_kernel<<<1, 32, 0, stream>>>(cnts);
    gru_transpose_kernel<<<dim3(GRU_T / 32, GRU_H / 32, GRU_B), dim3(32, 8), 0, stream>>>(inp, xA);

    const int nW = GRU_L * GRU_G3 * GRU_H;
    gru_cvt_bf16_kernel<<<(nW + 255) / 256, 256, 0, stream>>>(Wih, WihB, nW);
    gru_cvt_bf16_kernel<<<(nW + 255) / 256, 256, 0, stream>>>(Whh, WhhB, nW);
    gru_cvt_bf16_kernel<<<(GRU_O * GRU_H + 255) / 256, 256, 0, stream>>>(fcw, fcwB, GRU_O * GRU_H);

    for (int l = 0; l < GRU_L; ++l) {
        gru_gemm_bias_kernel<<<dim3(GRU_G3 / 128, GRU_M / 128), 256, 0, stream>>>(
            xA, WihB + (size_t)l * GRU_G3 * GRU_H, bih + (size_t)l * GRU_G3, xwbuf, GRU_G3);
        gru_recurrence_kernel<<<dim3(4, 4), 256, 0, stream>>>(
            xwbuf, WhhB + (size_t)l * GRU_G3 * GRU_H, bhh + (size_t)l * GRU_G3,
            xB, hbf, cnts + l * 4);
        float* tmp = xA; xA = xB; xB = tmp;
    }

    // FC: [32768][512] x [128][512]^T + fcb -> d_out [B,1,T,O]
    gru_gemm_bias_kernel<<<dim3(GRU_O / 128, GRU_M / 128), 256, 0, stream>>>(
        xA, fcwB, fcb, (float*)d_out, GRU_O);
}
